// EdgeBody_72086731096496
// MI455X (gfx1250) — matrix operand order repaired and checked by reading, not compiled, not run on hardware
//
#include <hip/hip_runtime.h>
#include <hip/hip_bf16.h>

typedef __attribute__((ext_vector_type(16))) _Float16 v16h;
typedef __attribute__((ext_vector_type(8)))  float    v8f;

#define N_NODES  20000
#define N_EDGES  320000
#define N_GRAPHS 64
#define DN  128
#define DE  128
#define DA  16
#define HIDW 256
#define KC1 528            // concat width for layer-1 K
#define KT1 17             // ceil(528/32) K-tiles, padded with zeros
#define NT1 16             // 256/16 N-tiles
#define KT2 8              // 256/32
#define NT2 8              // 128/16
#define AS  552            // LDS A-tile row stride in halfs (mult of 8)
#define W1F_CNT (KT1*NT1*32*16)   // 139264 halfs
#define W2F_CNT (KT2*NT2*32*16)   // 32768 halfs
#define ENC_NEG_INF 0x007FFFFFu   // encf(-inf)

// ---- order-preserving float<->uint keys for atomicMax on floats ----
__device__ __forceinline__ unsigned int encf(float f) {
    unsigned int u = __float_as_uint(f);
    return (u & 0x80000000u) ? ~u : (u | 0x80000000u);
}
__device__ __forceinline__ float decf(unsigned int k) {
    unsigned int u = (k & 0x80000000u) ? (k & 0x7FFFFFFFu) : ~k;
    return __uint_as_float(u);
}
__device__ __forceinline__ float wred(float v) {
    #pragma unroll
    for (int o = 16; o > 0; o >>= 1) v += __shfl_xor(v, o, 32);
    return v;
}

// ---------------- kernel 0: init accumulators ----------------
__global__ void init_kernel(unsigned* nmaxu, float* nsum, float* nacc,
                            unsigned* gmaxu, float* gsum, float* gacc) {
    int idx = blockIdx.x * blockDim.x + threadIdx.x;
    if (idx < N_NODES * DE) nacc[idx] = 0.f;
    if (idx < N_NODES) { nmaxu[idx] = ENC_NEG_INF; nsum[idx] = 0.f; }
    if (idx < N_GRAPHS * DE) gacc[idx] = 0.f;
    if (idx < N_GRAPHS) { gmaxu[idx] = ENC_NEG_INF; gsum[idx] = 0.f; }
}

// ---------------- kernel 1: convert W1/W2 to f16 B-fragment layout ----------
// B (32x16 KxN) per wave32: lanes 0-15 -> N=lane, K={0..7,16..23}; lanes 16-31 -> N=lane-16, K={8..15,24..31}
// fragment buffer: ((kt*NT + nt)*32 + lane)*16 + e16, k = kt*32 + (e16>>3)*16 + (lane>>4)*8 + (e16&7)
__global__ void prep_w_kernel(const float* __restrict__ W1, const float* __restrict__ W2,
                              _Float16* __restrict__ w1f, _Float16* __restrict__ w2f) {
    int idx = blockIdx.x * blockDim.x + threadIdx.x;
    if (idx < W1F_CNT) {
        int e16 = idx & 15, lane = (idx >> 4) & 31, nt = (idx >> 9) & 15, kt = idx >> 13;
        int k = kt * 32 + ((e16 >> 3) << 4) + ((lane >> 4) << 3) + (e16 & 7);
        int n = nt * 16 + (lane & 15);
        w1f[idx] = (_Float16)((k < KC1) ? W1[k * HIDW + n] : 0.f);
    }
    if (idx < W2F_CNT) {
        int e16 = idx & 15, lane = (idx >> 4) & 31, nt = (idx >> 9) & 7, kt = idx >> 12;
        int k = kt * 32 + ((e16 >> 3) << 4) + ((lane >> 4) << 3) + (e16 & 7);
        int n = nt * 16 + (lane & 15);
        w2f[idx] = (_Float16)W2[k * DE + n];
    }
}

// ---------------- kernel 2: aggregation logits + node max --------------------
__global__ __launch_bounds__(256) void agg_logit_kernel(
        const float* __restrict__ x, const float* __restrict__ hid,
        const float* __restrict__ attr, const int* __restrict__ rol,
        const int* __restrict__ col, const float* __restrict__ Wagg,
        const float* __restrict__ bagg, float* __restrict__ logits,
        unsigned* __restrict__ nmaxu) {
    int gid = blockIdx.x * blockDim.x + threadIdx.x;
    int lane = gid & 31, wv = gid >> 5;
    int nw = (gridDim.x * blockDim.x) >> 5;
    float ba = bagg[0];
    for (int e = wv; e < N_EDGES; e += nw) {
        int r = rol[e], c = col[e];
        const float* xr = x + (size_t)r * DN;
        const float* xc = x + (size_t)c * DN;
        const float* hr = hid + (size_t)e * DE;
        float part = 0.f;
        for (int j = lane; j < 400; j += 32) {
            float v;
            if (j < 128)      v = xr[j];
            else if (j < 256) v = xc[j - 128];
            else if (j < 272) v = attr[e * DA + j - 256];
            else              v = hr[j - 272];
            part += v * Wagg[j];
        }
        part = wred(part);
        if (lane == 0) {
            float lg = part + ba;
            logits[e] = lg;
            atomicMax(&nmaxu[c], encf(lg));
        }
    }
}

// ---------------- kernel 3: exp + scatter to node accumulators ---------------
__global__ __launch_bounds__(256) void node_acc_kernel(
        const float* __restrict__ hid, const int* __restrict__ col,
        const float* __restrict__ logits, const unsigned* __restrict__ nmaxu,
        float* __restrict__ nsum, float* __restrict__ nacc) {
    int gid = blockIdx.x * blockDim.x + threadIdx.x;
    int lane = gid & 31, wv = gid >> 5;
    int nw = (gridDim.x * blockDim.x) >> 5;
    for (int e = wv; e < N_EDGES; e += nw) {
        int c = col[e];
        float ew = __expf(logits[e] - decf(nmaxu[c]));
        if (lane == 0) atomicAdd(&nsum[c], ew);
        const float* hr = hid + (size_t)e * DE;
        for (int j = lane; j < DE; j += 32)
            atomicAdd(&nacc[(size_t)c * DE + j], ew * hr[j]);
    }
}

// ---------------- kernel 4: normalize node_feat ------------------------------
__global__ void node_norm_kernel(float* __restrict__ nacc, const float* __restrict__ nsum) {
    int idx = blockIdx.x * blockDim.x + threadIdx.x;
    if (idx < N_NODES * DE) nacc[idx] = nacc[idx] / (nsum[idx >> 7] + 1e-16f);
}

// ---------------- kernel 5: WMMA MLP + gate + readout logit ------------------
// block = 64 threads (2 waves), 32 edges; each wave owns a 16-edge M-tile.
__global__ __launch_bounds__(64) void mlp_gate_kernel(
        const float* __restrict__ x, const float* __restrict__ hid,
        const float* __restrict__ attr, const int* __restrict__ rol,
        const int* __restrict__ col, const int* __restrict__ batch,
        const float* __restrict__ nf,
        const _Float16* __restrict__ w1f, const float* __restrict__ b1,
        const _Float16* __restrict__ w2f, const float* __restrict__ b2,
        const float* __restrict__ Wupd, const float* __restrict__ bupd,
        const float* __restrict__ Wread, const float* __restrict__ bread,
        float* __restrict__ dout, float* __restrict__ rlog,
        unsigned* __restrict__ gmaxu) {
    __shared__ __align__(32) unsigned char smem[32 * AS * 2 + 32 * HIDW * 2]; // 51712 B
    _Float16* aT = (_Float16*)smem;                 // [32][AS]   layer-1 A (f16)
    _Float16* hT = (_Float16*)(smem + 32 * AS * 2); // [32][256]  hidden (f16)
    float*    cT = (float*)smem;                    // [32][128]  cand, aliases aT

    const int tid  = threadIdx.x;
    const int lane = tid & 31;
    const int wave = tid >> 5;
    const int half = lane >> 4;
    const int nloc = lane & 15;
    const int ebase = blockIdx.x * 32;

    // stage A tile: [xr | nf_r | xc | nf_c | attr | 0-pad] -> f16
    for (int idx = tid; idx < 32 * 544; idx += 64) {
        int row = idx / 544, cp = idx - row * 544;
        int e = ebase + row; if (e >= N_EDGES) e = N_EDGES - 1;
        int r = rol[e], c = col[e];
        float v;
        if (cp < 128)      v = x[(size_t)r * DN + cp];
        else if (cp < 256) v = nf[(size_t)r * DE + cp - 128];
        else if (cp < 384) v = x[(size_t)c * DN + cp - 256];
        else if (cp < 512) v = nf[(size_t)c * DE + cp - 384];
        else if (cp < 528) v = attr[e * DA + cp - 512];
        else               v = 0.f;
        aT[row * AS + cp] = (_Float16)v;
    }
    __syncthreads();

    const int arow = wave * 16 + nloc;   // A-fragment row (16-bit A layout: M = lane&15)

    // layer 1: H = relu(A @ W1 + b1)
    for (int nt = 0; nt < NT1; ++nt) {
        float bias = b1[nt * 16 + nloc];
        v8f acc = {bias, bias, bias, bias, bias, bias, bias, bias};
        for (int kt = 0; kt < KT1; ++kt) {
            v16h a;  // lanes 0-15: K = kt*32 + {0..7,16..23}; lanes 16-31: +8
            const _Float16* ap = aT + arow * AS + kt * 32 + half * 8;
            ((uint4*)&a)[0] = *(const uint4*)ap;
            ((uint4*)&a)[1] = *(const uint4*)(ap + 16);
            v16h b = *(const v16h*)(w1f + (size_t)((kt * NT1 + nt) * 32 + lane) * 16);
            acc = __builtin_amdgcn_wmma_f32_16x16x32_f16(false, a, false, b,
                                                         (short)0, acc, false, false);
        }
        #pragma unroll
        for (int v = 0; v < 8; ++v) {   // C layout: row = v + 8*half, col = lane&15
            int m = v + (half << 3);
            hT[(wave * 16 + m) * HIDW + nt * 16 + nloc] = (_Float16)fmaxf(acc[v], 0.f);
        }
    }
    __syncthreads();   // hT complete; aT region reusable as cT

    // layer 2: cand = H @ W2 + b2
    for (int nt = 0; nt < NT2; ++nt) {
        float bias = b2[nt * 16 + nloc];
        v8f acc = {bias, bias, bias, bias, bias, bias, bias, bias};
        for (int kt = 0; kt < KT2; ++kt) {
            v16h a;
            const _Float16* ap = hT + arow * HIDW + kt * 32 + half * 8;
            ((uint4*)&a)[0] = *(const uint4*)ap;
            ((uint4*)&a)[1] = *(const uint4*)(ap + 16);
            v16h b = *(const v16h*)(w2f + (size_t)((kt * NT2 + nt) * 32 + lane) * 16);
            acc = __builtin_amdgcn_wmma_f32_16x16x32_f16(false, a, false, b,
                                                         (short)0, acc, false, false);
        }
        #pragma unroll
        for (int v = 0; v < 8; ++v) {
            int m = v + (half << 3);
            cT[(wave * 16 + m) * DE + nt * 16 + nloc] = acc[v];
        }
    }
    __syncthreads();

    // gate + edge_feat + readout logit (wave cooperates per edge)
    float bu = bupd[0], br = bread[0];
    for (int m = 0; m < 16; ++m) {
        int e = ebase + wave * 16 + m;
        if (e >= N_EDGES) break;                 // uniform per wave
        int r = rol[e], c = col[e];
        const float* xr   = x + (size_t)r * DN;
        const float* xc   = x + (size_t)c * DN;
        const float* hrow = hid + (size_t)e * DE;
        const float* crow = cT + (wave * 16 + m) * DE;

        float part = 0.f;                        // 528-wide gate dot
        for (int j = lane; j < 528; j += 32) {
            float v;
            if (j < 128)      v = xr[j];
            else if (j < 256) v = xc[j - 128];
            else if (j < 272) v = attr[e * DA + j - 256];
            else if (j < 400) v = hrow[j - 272];
            else              v = crow[j - 400];
            part += v * Wupd[j];
        }
        part = wred(part);
        float g = 1.f / (1.f + __expf(-(part + bu)));

        float rp = 0.f;                          // 400-wide readout dot, fused with output
        for (int j = lane; j < 128; j += 32) {
            float ef = g * crow[j] + (1.f - g) * hrow[j];
            dout[(size_t)e * DE + j] = ef;
            rp += ef * Wread[256 + j];
            rp += xr[j] * Wread[j];
            rp += xc[j] * Wread[128 + j];
        }
        if (lane < 16) rp += attr[e * DA + lane] * Wread[384 + lane];
        rp = wred(rp);
        if (lane == 0) {
            float lg = rp + br;
            rlog[e] = lg;
            atomicMax(&gmaxu[batch[r]], encf(lg));
        }
    }
}

// ---------------- kernel 6: graph softmax accumulate (LDS-staged) ------------
__global__ __launch_bounds__(256) void graph_acc_kernel(
        const int* __restrict__ rol, const int* __restrict__ batch,
        const float* __restrict__ rlog, const unsigned* __restrict__ gmaxu,
        const float* __restrict__ dout, float* __restrict__ gsum,
        float* __restrict__ gacc) {
    __shared__ float gsL[N_GRAPHS];
    __shared__ float gaL[N_GRAPHS * DE];
    int tid = threadIdx.x, lane = tid & 31, wave = tid >> 5;
    for (int i = tid; i < N_GRAPHS; i += 256) gsL[i] = 0.f;
    for (int i = tid; i < N_GRAPHS * DE; i += 256) gaL[i] = 0.f;
    __syncthreads();
    int per = (N_EDGES + gridDim.x - 1) / gridDim.x;
    int e0 = blockIdx.x * per;
    int e1 = min(e0 + per, N_EDGES);
    for (int e = e0 + wave; e < e1; e += 8) {
        int gix = batch[rol[e]];
        float ew = __expf(rlog[e] - decf(gmaxu[gix]));
        if (lane == 0) atomicAdd(&gsL[gix], ew);
        for (int j = lane; j < DE; j += 32)
            atomicAdd(&gaL[gix * DE + j], ew * dout[(size_t)e * DE + j]);
    }
    __syncthreads();
    for (int i = tid; i < N_GRAPHS; i += 256)
        if (gsL[i] != 0.f) atomicAdd(&gsum[i], gsL[i]);
    for (int i = tid; i < N_GRAPHS * DE; i += 256)
        if (gaL[i] != 0.f) atomicAdd(&gacc[i], gaL[i]);
}

// ---------------- kernel 7: confidence ---------------------------------------
__global__ __launch_bounds__(128) void conf_kernel(
        const float* __restrict__ gacc, const float* __restrict__ gsum,
        const float* __restrict__ Wsc, const float* __restrict__ bsc,
        float* __restrict__ dout) {
    __shared__ float red[128];
    int g = blockIdx.x, tid = threadIdx.x;
    float denom = gsum[g] + 1e-16f;
    red[tid] = (gacc[g * DE + tid] / denom) * Wsc[tid];
    __syncthreads();
    for (int s = 64; s > 0; s >>= 1) {
        if (tid < s) red[tid] += red[tid + s];
        __syncthreads();
    }
    if (tid == 0)
        dout[(size_t)N_EDGES * DE + g] = 1.f / (1.f + __expf(-(red[0] + bsc[0])));
}

extern "C" void kernel_launch(void* const* d_in, const int* in_sizes, int n_in,
                              void* d_out, int out_size, void* d_ws, size_t ws_size,
                              hipStream_t stream) {
    const float* x     = (const float*)d_in[0];
    const float* hid   = (const float*)d_in[1];
    const float* attr  = (const float*)d_in[2];
    const int*   ei    = (const int*)d_in[3];
    const int*   batch = (const int*)d_in[4];
    /* d_in[5] = num_graphs (compile-time N_GRAPHS) */
    const float* Wagg  = (const float*)d_in[6];
    const float* bagg  = (const float*)d_in[7];
    const float* W1    = (const float*)d_in[8];
    const float* b1    = (const float*)d_in[9];
    const float* W2    = (const float*)d_in[10];
    const float* b2    = (const float*)d_in[11];
    const float* Wupd  = (const float*)d_in[12];
    const float* bupd  = (const float*)d_in[13];
    const float* Wread = (const float*)d_in[14];
    const float* bread = (const float*)d_in[15];
    const float* Wsc   = (const float*)d_in[16];
    const float* bsc   = (const float*)d_in[17];
    const int* rol = ei;
    const int* col = ei + N_EDGES;

    char* ws = (char*)d_ws;
    size_t off = 0;
    auto bump = [&](size_t bytes) {
        size_t o = off; off = (off + bytes + 255) & ~(size_t)255; return o;
    };
    float*    logits = (float*)(ws + bump((size_t)N_EDGES * 4));      // reused as rlog
    unsigned* nmaxu  = (unsigned*)(ws + bump((size_t)N_NODES * 4));
    float*    nsum   = (float*)(ws + bump((size_t)N_NODES * 4));
    float*    nacc   = (float*)(ws + bump((size_t)N_NODES * DE * 4)); // becomes node_feat
    unsigned* gmaxu  = (unsigned*)(ws + bump((size_t)N_GRAPHS * 4));
    float*    gsum   = (float*)(ws + bump((size_t)N_GRAPHS * 4));
    float*    gacc   = (float*)(ws + bump((size_t)N_GRAPHS * DE * 4));
    _Float16* w1f    = (_Float16*)(ws + bump((size_t)W1F_CNT * 2));
    _Float16* w2f    = (_Float16*)(ws + bump((size_t)W2F_CNT * 2));
    float* dout = (float*)d_out;

    init_kernel<<<(N_NODES * DE + 255) / 256, 256, 0, stream>>>(
        nmaxu, nsum, nacc, gmaxu, gsum, gacc);
    prep_w_kernel<<<(W1F_CNT + 255) / 256, 256, 0, stream>>>(W1, W2, w1f, w2f);
    agg_logit_kernel<<<2048, 256, 0, stream>>>(x, hid, attr, rol, col, Wagg, bagg,
                                               logits, nmaxu);
    node_acc_kernel<<<2048, 256, 0, stream>>>(hid, col, logits, nmaxu, nsum, nacc);
    node_norm_kernel<<<(N_NODES * DE + 255) / 256, 256, 0, stream>>>(nacc, nsum);
    mlp_gate_kernel<<<(N_EDGES + 31) / 32, 64, 0, stream>>>(
        x, hid, attr, rol, col, batch, nacc, w1f, b1, w2f, b2,
        Wupd, bupd, Wread, bread, dout, logits, gmaxu);
    graph_acc_kernel<<<512, 256, 0, stream>>>(rol, batch, logits, gmaxu, dout, gsum, gacc);
    conf_kernel<<<N_GRAPHS, 128, 0, stream>>>(gacc, gsum, Wsc, bsc, dout);
}
